// ODEPredictor_65051574665510
// MI455X (gfx1250) — matrix operand order repaired and checked by reading, not compiled, not run on hardware
//
#include <hip/hip_runtime.h>
#include <math.h>
#include <stdint.h>

// ---------------------------------------------------------------------------
// Problem dims (from reference)
// ---------------------------------------------------------------------------
constexpr int Bsz = 2048, Lsz = 96, Psz = 96, CINsz = 8, Nsz = 64, HDsz = 64,
              PHsz = 8, Ksz = 3;
constexpr int NTILES_B = Bsz / 16;  // 128 batch tiles

typedef __attribute__((ext_vector_type(16))) _Float16 v16h;
typedef __attribute__((ext_vector_type(8)))  _Float16 v8h;
typedef __attribute__((ext_vector_type(8)))  float    v8f;
typedef __attribute__((ext_vector_type(4)))  float    v4f;
typedef __attribute__((ext_vector_type(4)))  int      v4i;

#define LDS_FENCE() __asm__ __volatile__("" ::: "memory")

enum { ACT_NONE = 0, ACT_SILU = 1, ACT_TANH = 2 };

// ---------------------------------------------------------------------------
// Async Global->LDS DMA (CDNA5, tracked by ASYNCcnt).  Guarded so either
// toolchain still compiles; falls back to a VGPR round-trip copy.
// Builtin signature (from probe diagnostic): non-const v4i pointers.
// ---------------------------------------------------------------------------
#if defined(__has_builtin)
#if __has_builtin(__builtin_amdgcn_global_load_async_to_lds_b128) && \
    __has_builtin(__builtin_amdgcn_s_wait_asynccnt)
#define HAVE_ASYNC_LDS 1
#endif
#endif
#ifndef HAVE_ASYNC_LDS
#define HAVE_ASYNC_LDS 0
#endif

#if HAVE_ASYNC_LDS
typedef __attribute__((address_space(1))) v4i* g4_ptr_t;
typedef __attribute__((address_space(3))) v4i* l4_ptr_t;
#endif

__device__ __forceinline__ void async_g2l_b128(const void* g, void* l) {
#if HAVE_ASYNC_LDS
  // AS3 pointers are 32-bit; low 32 bits of a generic LDS address are the
  // LDS byte offset (ISA aperture rule), so the integer cast is exact.
  __builtin_amdgcn_global_load_async_to_lds_b128(
      (g4_ptr_t)(uintptr_t)g, (l4_ptr_t)(uint32_t)(uintptr_t)l, 0, 0);
#else
  *(v4f*)l = *(const v4f*)g;
#endif
}

__device__ __forceinline__ void async_g2l_wait() {
#if HAVE_ASYNC_LDS
  __builtin_amdgcn_s_wait_asynccnt(0);
#endif
}

// ---------------------------------------------------------------------------
// WMMA helpers (wave32, V_WMMA_F32_16X16X32_F16)
// ---------------------------------------------------------------------------
__device__ __forceinline__ v8f wmma32(v16h a, v16h b, v8f c) {
  return __builtin_amdgcn_wmma_f32_16x16x32_f16(
      /*neg_a=*/false, a, /*neg_b=*/false, b,
      /*c_mod=*/(short)0, c, /*reuse_a=*/false, /*reuse_b=*/false);
}

// A-fragment (16xK tile): halves 0-15 => K k1..k1+15 (contiguous, matches
// bfrag), with k1 = kbase + (laneHi ? 16 : 0).
__device__ __forceinline__ v16h afrag(const _Float16* act, int stride, int kbase) {
  const int lane = threadIdx.x & 31;
  const _Float16* row = act + (lane & 15) * stride + kbase + ((lane >> 4) << 4);
  const v8h lo = *(const v8h*)(row);
  const v8h hi = *(const v8h*)(row + 8);
  v16h r;
#pragma unroll
  for (int i = 0; i < 8; ++i) { r[i] = lo[i]; r[i + 8] = hi[i]; }
  return r;
}

// B-fragment for y = x @ W^T : lane holds column n = lane%16, 16 contiguous K
// (lanes 0-15: kbase..+15, lanes 16-31: kbase+16..+31). Zero-pads n>=Nout,
// k>=Kw so K=8 / N=8 layers run on the same instruction.
__device__ __forceinline__ v16h bfrag(const _Float16* __restrict__ W, int Nout,
                                      int Kw, int ntile, int kbase) {
  const int lane = threadIdx.x & 31;
  const int n = ntile * 16 + (lane & 15);
  const int k0 = kbase + ((lane >> 4) << 4);
  v16h bv;
  if (n < Nout && (k0 + 16) <= Kw) {
    bv = *(const v16h*)(W + (size_t)n * Kw + k0);
  } else {
#pragma unroll
    for (int i = 0; i < 16; ++i) {
      const int k = k0 + i;
      bv[i] = (n < Nout && k < Kw) ? W[(size_t)n * Kw + k] : (_Float16)0.f;
    }
  }
  return bv;
}

// Fast activations: v_rcp_f32 instead of IEEE-div Newton iterations.
__device__ __forceinline__ float fast_sig(float v) {
  return __builtin_amdgcn_rcpf(1.f + __expf(-v));
}
__device__ __forceinline__ float apply_act(float v, int mode) {
  if (mode == ACT_SILU) return v * fast_sig(v);
  if (mode == ACT_TANH)
    return 1.f - 2.f * __builtin_amdgcn_rcpf(__expf(2.f * v) + 1.f);
  return v;
}

// C/D layout: VGPR r -> row (lane<16 ? r : r+8), col n = lane%16.
__device__ __forceinline__ void store_d_f16(v8f acc, const float* bias, int ntile,
                                            int Nout, _Float16* buf, int stride,
                                            int mode) {
  const int lane = threadIdx.x & 31;
  const int n = ntile * 16 + (lane & 15);
  if (n < Nout) {
    const int mb = (lane >> 4) * 8;
    const float bz = bias[n];
#pragma unroll
    for (int r = 0; r < 8; ++r)
      buf[(mb + r) * stride + n] = (_Float16)apply_act(acc[r] + bz, mode);
  }
}

__device__ __forceinline__ void store_d_f32(v8f acc, const float* bias, int ntile,
                                            int Nout, float* buf, int stride,
                                            int mode) {
  const int lane = threadIdx.x & 31;
  const int n = ntile * 16 + (lane & 15);
  if (n < Nout) {
    const int mb = (lane >> 4) * 8;
    const float bz = bias[n];
#pragma unroll
    for (int r = 0; r < 8; ++r)
      buf[(mb + r) * stride + n] = apply_act(acc[r] + bz, mode);
  }
}

// Linear layer on a 16-row tile. A-fragments preloaded once (reused over all
// N-tiles); actIn/actOut never alias in our call sites.
__device__ __forceinline__ void mlp_layer(const _Float16* __restrict__ actIn,
                                          int sIn, int Kin,
                                          const _Float16* __restrict__ W, int Kw,
                                          const float* __restrict__ bias,
                                          int Nout, _Float16* __restrict__ actOut,
                                          int sOut, int mode) {
  v16h a[2];
  const int nk = Kin >> 5;
#pragma unroll
  for (int i = 0; i < 2; ++i)
    if (i < nk) a[i] = afrag(actIn, sIn, i * 32);
  const int ntiles = (Nout + 15) >> 4;
  for (int nt = 0; nt < ntiles; ++nt) {
    v8f acc = {0.f, 0.f, 0.f, 0.f, 0.f, 0.f, 0.f, 0.f};
#pragma unroll
    for (int i = 0; i < 2; ++i)
      if (i < nk) acc = wmma32(a[i], bfrag(W, Nout, Kw, nt, i * 32), acc);
    store_d_f16(acc, bias, nt, Nout, actOut, sOut, mode);
  }
}

__device__ __forceinline__ void mlp_layer_f32(const _Float16* __restrict__ actIn,
                                              int sIn, int Kin,
                                              const _Float16* __restrict__ W,
                                              int Kw,
                                              const float* __restrict__ bias,
                                              int Nout, float* __restrict__ outBuf,
                                              int sOut, int mode) {
  v16h a[2];
  const int nk = Kin >> 5;
#pragma unroll
  for (int i = 0; i < 2; ++i)
    if (i < nk) a[i] = afrag(actIn, sIn, i * 32);
  const int ntiles = (Nout + 15) >> 4;
  for (int nt = 0; nt < ntiles; ++nt) {
    v8f acc = {0.f, 0.f, 0.f, 0.f, 0.f, 0.f, 0.f, 0.f};
#pragma unroll
    for (int i = 0; i < 2; ++i)
      if (i < nk) acc = wmma32(a[i], bfrag(W, Nout, Kw, nt, i * 32), acc);
    store_d_f32(acc, bias, nt, Nout, outBuf, sOut, mode);
  }
}

// ---------------------------------------------------------------------------
// Utility kernels
// ---------------------------------------------------------------------------
__global__ void cvt_f16_kernel(const float* __restrict__ in,
                               _Float16* __restrict__ out, int n) {
  int i = blockIdx.x * blockDim.x + threadIdx.x;
  if (i < n) out[i] = (_Float16)in[i];
}

__global__ void copy_f32_kernel(const float* __restrict__ in,
                                float* __restrict__ out, int n) {
  int i = blockIdx.x * blockDim.x + threadIdx.x;
  if (i < n) out[i] = in[i];
}

// ---------------------------------------------------------------------------
// x_proj MLP over all B*L tokens: (8 ->64 ->64 ->64 ->64 ->64), SiLU between.
// ---------------------------------------------------------------------------
__global__ __launch_bounds__(128) void xproj_kernel(
    const float* __restrict__ x, const _Float16* w0, const float* b0,
    const _Float16* w1, const float* b1, const _Float16* w2, const float* b2,
    const _Float16* w3, const float* b3, const _Float16* w4, const float* b4,
    float* __restrict__ xold) {
  __shared__ __align__(32) _Float16 A0[4][16 * 64];
  __shared__ __align__(32) _Float16 A1[4][16 * 64];
  const int wave = threadIdx.x >> 5, lane = threadIdx.x & 31;
  const size_t R = (size_t)(blockIdx.x * 4 + wave) * 16;
  _Float16* a0 = A0[wave];
  _Float16* a1 = A1[wave];

  for (int i = lane; i < 16 * 32; i += 32) {
    int m = i >> 5, c = i & 31;
    a0[m * 64 + c] = (c < CINsz) ? (_Float16)x[(R + m) * CINsz + c] : (_Float16)0.f;
  }
  LDS_FENCE();
  mlp_layer(a0, 64, 32, w0, CINsz, b0, HDsz, a1, 64, ACT_SILU);
  mlp_layer(a1, 64, 64, w1, HDsz, b1, HDsz, a0, 64, ACT_SILU);
  mlp_layer(a0, 64, 64, w2, HDsz, b2, HDsz, a1, 64, ACT_SILU);
  mlp_layer(a1, 64, 64, w3, HDsz, b3, HDsz, a0, 64, ACT_SILU);
  mlp_layer_f32(a0, 64, 64, w4, HDsz, b4, Nsz, xold + R * Nsz, Nsz, ACT_NONE);
}

// ---------------------------------------------------------------------------
// Depthwise dilated conv + SiLU + comp projection (6144 -> 8).
// x_old tile staged into LDS via async Global->LDS DMA (ASYNCcnt).
// ---------------------------------------------------------------------------
__global__ __launch_bounds__(256) void conv_comp_kernel(
    const float* __restrict__ xold, const float* __restrict__ cw,
    const float* __restrict__ cb, const float* __restrict__ compw,
    const float* __restrict__ compb, float* __restrict__ a0buf) {
  __shared__ __align__(16) float xo[Lsz * Nsz];
  __shared__ float red[PHsz];
  const int b = blockIdx.x, tid = threadIdx.x;
#pragma unroll
  for (int i = 0; i < 6; ++i) {  // 6144 floats / (256 threads * 4) = 6
    const int idx = (tid + i * 256) * 4;
    async_g2l_b128(xold + (size_t)b * (Lsz * Nsz) + idx, xo + idx);
  }
  if (tid < PHsz) red[tid] = 0.f;
  async_g2l_wait();
  __syncthreads();

  float part[PHsz];
#pragma unroll
  for (int q = 0; q < PHsz; ++q) part[q] = 0.f;

#pragma unroll
  for (int i = 0; i < 24; ++i) {
    const int e = tid + i * 256;
    __builtin_prefetch(compw + e + 1024, 0, 1);  // global_prefetch_b8
    const int l = e >> 6, n = e & 63;
    float s = cb[n];
#pragma unroll
    for (int k = 0; k < Ksz; ++k) {
      const int ll = l + 2 * k;
      if (ll < Lsz) s += cw[n * Ksz + k] * xo[ll * Nsz + n];
    }
    const float hv = s * fast_sig(s);
#pragma unroll
    for (int q = 0; q < PHsz; ++q) part[q] += compw[q * (Lsz * Nsz) + e] * hv;
  }
#pragma unroll
  for (int q = 0; q < PHsz; ++q) atomicAdd(&red[q], part[q]);
  __syncthreads();
  if (tid < PHsz) a0buf[b * PHsz + tid] = red[tid] + compb[tid];
}

// ---------------------------------------------------------------------------
// Householder direction v = normalize( xpp1( silu( xpp0( z0 ) ) ) )
// ---------------------------------------------------------------------------
__global__ __launch_bounds__(128) void v_kernel(const float* __restrict__ xold,
                                                const _Float16* w0, const float* b0,
                                                const _Float16* w1, const float* b1,
                                                float* __restrict__ vglob) {
  __shared__ __align__(32) _Float16 A0[4][16 * 64];
  __shared__ __align__(32) _Float16 A1[4][16 * 64];
  __shared__ float VT[4][16 * 64];
  const int wave = threadIdx.x >> 5, lane = threadIdx.x & 31;
  const int bb = (blockIdx.x * 4 + wave) * 16;
  _Float16* a0 = A0[wave];
  _Float16* a1 = A1[wave];
  float* vt = VT[wave];

  for (int i = lane; i < 16 * 64; i += 32) {
    int m = i >> 6, n = i & 63;
    a0[i] = (_Float16)xold[((size_t)(bb + m) * Lsz + (Lsz - 1)) * Nsz + n];
  }
  LDS_FENCE();
  mlp_layer(a0, 64, 64, w0, HDsz, b0, HDsz, a1, 64, ACT_SILU);
  mlp_layer_f32(a1, 64, 64, w1, HDsz, b1, Nsz, vt, 64, ACT_NONE);
  LDS_FENCE();

  const int m = lane & 15, n0 = (lane >> 4) * 32;
  float ss = 0.f;
#pragma unroll
  for (int j = 0; j < 32; ++j) { float t = vt[m * 64 + n0 + j]; ss += t * t; }
  ss += __shfl_xor(ss, 16);
  const float inv = 1.f / (sqrtf(ss) + 1e-8f);
#pragma unroll
  for (int j = 0; j < 32; ++j)
    vglob[(size_t)(bb + m) * Nsz + n0 + j] = vt[m * 64 + n0 + j] * inv;
}

// ---------------------------------------------------------------------------
// Sequential recurrence only (11 WMMA / step), streaming the (a,z) trajectory
// to workspace as f16.  One wave = 16 batch rows; weights staged in LDS.
// ---------------------------------------------------------------------------
__global__ __launch_bounds__(32) void scan_kernel(
    const float* __restrict__ a0g, const float* __restrict__ xold,
    const float* __restrict__ vglob, const _Float16* __restrict__ nw0,
    const float* __restrict__ nb0, const _Float16* __restrict__ nw1,
    const float* __restrict__ nb1, const _Float16* __restrict__ low0,
    const float* __restrict__ lob0, const _Float16* __restrict__ low1,
    const float* __restrict__ lob1, const float* __restrict__ spanA,
    _Float16* __restrict__ atraj, _Float16* __restrict__ ztraj) {
  __shared__ __align__(32) _Float16 aA[16 * 32];
  __shared__ __align__(32) _Float16 h8[16 * 32];
  __shared__ __align__(32) _Float16 hA[16 * 64];
  __shared__ __align__(32) _Float16 wn0[512], wn1[512], wl0[64], wl1[512];
  __shared__ float af[16 * 8], daf[16 * 8];
  __shared__ float zf[16 * 64], vf[16 * 64], df[16 * 64];

  const int lane = threadIdx.x & 31;
  const int tb = blockIdx.x;  // batch tile
  const int bb = tb * 16;
  float dt = spanA[0];
  dt = fminf(fmaxf(dt, 1e-8f), 7.0f);

  for (int i = lane; i < 512; i += 32) {
    wn0[i] = nw0[i]; wn1[i] = nw1[i]; wl1[i] = low1[i];
  }
  for (int i = lane; i < 64; i += 32) wl0[i] = low0[i];
  for (int i = lane; i < 16 * 32; i += 32) {
    aA[i] = (_Float16)0.f;
    h8[i] = (_Float16)0.f;
  }
  for (int i = lane; i < 16 * 8; i += 32) {
    const int m = i >> 3, c = i & 7;
    const float v = a0g[(bb + m) * PHsz + c];
    af[i] = v;
    aA[m * 32 + c] = (_Float16)v;
  }
  for (int i = lane; i < 16 * 64; i += 32) {
    const int m = i >> 6, n = i & 63;
    const float z = xold[((size_t)(bb + m) * Lsz + (Lsz - 1)) * Nsz + n];
    zf[i] = z;
    vf[i] = vglob[(size_t)(bb + m) * Nsz + n];
  }
  LDS_FENCE();

  for (int p = 0; p < Psz; ++p) {
    _Float16* zt = ztraj + ((size_t)p * NTILES_B + tb) * 16 * 64;
    _Float16* at = atraj + ((size_t)p * NTILES_B + tb) * 16 * 8;

    // da = tanh( net1( silu( net0(a) ) ) )
    mlp_layer(aA, 32, 32, wn0, PHsz, nb0, HDsz, hA, 64, ACT_SILU);
    mlp_layer_f32(hA, 64, 64, wn1, HDsz, nb1, PHsz, daf, 8, ACT_TANH);
    // d = lo1( silu( lo0(a) ) )
    mlp_layer(aA, 32, 32, wl0, PHsz, lob0, PHsz, h8, 32, ACT_SILU);
    mlp_layer_f32(h8, 32, 32, wl1, PHsz, lob1, Nsz, df, 64, ACT_NONE);
    LDS_FENCE();

    // Householder reflection + Euler update (2 lanes per row)
    {
      const int m = lane & 15, n0 = (lane >> 4) * 32;
      const float* vr = vf + m * 64 + n0;
      float* zr = zf + m * 64 + n0;
      const float* dr = df + m * 64 + n0;
      float vz = 0.f;
#pragma unroll
      for (int j = 0; j < 32; ++j) vz += vr[j] * zr[j];
      vz += __shfl_xor(vz, 16);
      float wl[32], vw = 0.f;
#pragma unroll
      for (int j = 0; j < 32; ++j) {
        const float Hz = zr[j] - 2.f * vz * vr[j];
        const float w = dr[j] * Hz;
        wl[j] = w;
        vw += vr[j] * w;
      }
      vw += __shfl_xor(vw, 16);
#pragma unroll
      for (int j = 0; j < 32; ++j) {
        const float dz = wl[j] - 2.f * vw * vr[j];
        const float zn = zr[j] + dt * dz;
        zr[j] = zn;
        zt[m * 64 + n0 + j] = (_Float16)zn;  // trajectory out (global)
      }
#pragma unroll
      for (int i = 0; i < 4; ++i) {
        const int e = lane * 4 + i;
        const int m2 = e >> 3, c = e & 7;
        const float an = af[e] + dt * daf[e];
        af[e] = an;
        aA[m2 * 32 + c] = (_Float16)an;
        at[e] = (_Float16)an;  // trajectory out (global)
      }
    }
    LDS_FENCE();
  }
}

// ---------------------------------------------------------------------------
// Fully-parallel decoders over the (P x B) trajectory: 43 WMMA per 16-row
// tile.  z tiles feed WMMA A-operands DIRECTLY from global memory.
// ---------------------------------------------------------------------------
__global__ __launch_bounds__(64) void decoder_kernel(
    const _Float16* __restrict__ atraj, const _Float16* __restrict__ ztraj,
    const _Float16* __restrict__ lcw0, const float* __restrict__ lcb0,
    const _Float16* __restrict__ lcw1, const float* __restrict__ lcb1,
    const _Float16* __restrict__ lzw0, const float* __restrict__ lzb0,
    const _Float16* __restrict__ lzw1, const float* __restrict__ lzb1,
    float* __restrict__ xpred) {
  __shared__ __align__(32) _Float16 AA[2][16 * 32];
  __shared__ __align__(32) _Float16 H8[2][16 * 32];
  __shared__ __align__(32) _Float16 HA[2][16 * 64];
  __shared__ __align__(32) _Float16 CB[2][16 * 512];
  __shared__ float AZ[2][16 * 8];

  const int wave = threadIdx.x >> 5, lane = threadIdx.x & 31;
  const int t = blockIdx.x * 2 + wave;  // tile id in [0, P*NTILES_B)
  const int p = t / NTILES_B, tb = t % NTILES_B;
  const int bb = tb * 16;
  _Float16* aA = AA[wave];
  _Float16* h8 = H8[wave];
  _Float16* hA = HA[wave];
  _Float16* Cb = CB[wave];
  float* azf = AZ[wave];

  const _Float16* at = atraj + ((size_t)p * NTILES_B + tb) * 16 * 8;
  const _Float16* zt = ztraj + ((size_t)p * NTILES_B + tb) * 16 * 64;

  for (int i = lane; i < 16 * 32; i += 32) {
    const int c = i & 31;
    aA[i] = (c < PHsz) ? at[(i >> 5) * 8 + c] : (_Float16)0.f;
    h8[i] = (_Float16)0.f;
  }
  LDS_FENCE();

  // C = lc1( silu( lc0(a) ) ) : 16x512 rows
  mlp_layer(aA, 32, 32, lcw0, PHsz, lcb0, PHsz, h8, 32, ACT_SILU);
  mlp_layer(h8, 32, 32, lcw1, PHsz, lcb1, CINsz * Nsz, Cb, 512, ACT_NONE);
  // added_z = lz1( silu( lz0(z) ) ) : z straight from global trajectory
  mlp_layer(zt, 64, 64, lzw0, Nsz, lzb0, Nsz, hA, 64, ACT_SILU);
  mlp_layer_f32(hA, 64, 64, lzw1, Nsz, lzb1, CINsz, azf, 8, ACT_NONE);
  LDS_FENCE();

  // y_hat[c] = sum_n C[c*64+n] * z[n];  x_pred = y_hat + added_z
#pragma unroll
  for (int i = 0; i < 4; ++i) {
    const int e = lane * 4 + i;
    const int m = e >> 3, c = e & 7;
    float y = azf[e];
    const _Float16* Crow = Cb + m * 512 + c * 64;
    const _Float16* zrow = zt + m * 64;
#pragma unroll
    for (int n8 = 0; n8 < 8; ++n8) {
      const v8h cc = *(const v8h*)(Crow + n8 * 8);
      const v8h zz = *(const v8h*)(zrow + n8 * 8);
#pragma unroll
      for (int j = 0; j < 8; ++j) y += (float)cc[j] * (float)zz[j];
    }
    xpred[((size_t)(bb + m) * Psz + p) * CINsz + c] = y;
  }
}

// ---------------------------------------------------------------------------
// Launch
// ---------------------------------------------------------------------------
extern "C" void kernel_launch(void* const* d_in, const int* in_sizes, int n_in,
                              void* d_out, int out_size, void* d_ws,
                              size_t ws_size, hipStream_t stream) {
  (void)in_sizes; (void)n_in; (void)out_size; (void)ws_size;
  const float* x      = (const float*)d_in[0];
  const float* xp_w0  = (const float*)d_in[1];  const float* xp_b0 = (const float*)d_in[2];
  const float* xp_w1  = (const float*)d_in[3];  const float* xp_b1 = (const float*)d_in[4];
  const float* xp_w2  = (const float*)d_in[5];  const float* xp_b2 = (const float*)d_in[6];
  const float* xp_w3  = (const float*)d_in[7];  const float* xp_b3 = (const float*)d_in[8];
  const float* xp_w4  = (const float*)d_in[9];  const float* xp_b4 = (const float*)d_in[10];
  const float* xpp_w0 = (const float*)d_in[11]; const float* xpp_b0 = (const float*)d_in[12];
  const float* xpp_w1 = (const float*)d_in[13]; const float* xpp_b1 = (const float*)d_in[14];
  const float* conv_w = (const float*)d_in[15]; const float* conv_b = (const float*)d_in[16];
  const float* span_A = (const float*)d_in[17];
  const float* comp_w = (const float*)d_in[18]; const float* comp_b = (const float*)d_in[19];
  const float* net_w0 = (const float*)d_in[20]; const float* net_b0 = (const float*)d_in[21];
  const float* net_w1 = (const float*)d_in[22]; const float* net_b1 = (const float*)d_in[23];
  const float* lo_w0  = (const float*)d_in[24]; const float* lo_b0 = (const float*)d_in[25];
  const float* lo_w1  = (const float*)d_in[26]; const float* lo_b1 = (const float*)d_in[27];
  const float* lc_w0  = (const float*)d_in[28]; const float* lc_b0 = (const float*)d_in[29];
  const float* lc_w1  = (const float*)d_in[30]; const float* lc_b1 = (const float*)d_in[31];
  const float* lz_w0  = (const float*)d_in[32]; const float* lz_b0 = (const float*)d_in[33];
  const float* lz_w1  = (const float*)d_in[34]; const float* lz_b1 = (const float*)d_in[35];

  float* out = (float*)d_out;

  // workspace layout
  float* xold = (float*)d_ws;                    // B*L*N f32
  float* a0b  = xold + (size_t)Bsz * Lsz * Nsz;  // B*8
  float* vbuf = a0b + (size_t)Bsz * PHsz;        // B*64
  _Float16* wh = (_Float16*)(vbuf + (size_t)Bsz * Nsz);
  _Float16* h_xpw0 = wh + 0;      // (64,8)
  _Float16* h_xpw1 = wh + 512;    // (64,64)
  _Float16* h_xpw2 = wh + 4608;
  _Float16* h_xpw3 = wh + 8704;
  _Float16* h_xpw4 = wh + 12800;
  _Float16* h_xppw0 = wh + 16896;
  _Float16* h_xppw1 = wh + 20992;
  _Float16* h_nw0  = wh + 25088;  // (64,8)
  _Float16* h_nw1  = wh + 25600;  // (8,64)
  _Float16* h_low0 = wh + 26112;  // (8,8)
  _Float16* h_low1 = wh + 26176;  // (64,8)
  _Float16* h_lcw0 = wh + 26688;  // (8,8)
  _Float16* h_lcw1 = wh + 26752;  // (512,8)
  _Float16* h_lzw0 = wh + 30848;  // (64,64)
  _Float16* h_lzw1 = wh + 34944;  // (8,64)
  _Float16* atraj  = wh + 35456;                        // P*B*8  f16
  _Float16* ztraj  = atraj + (size_t)Psz * Bsz * PHsz;  // P*B*64 f16

  auto cv = [&](const float* s, _Float16* d, int n) {
    cvt_f16_kernel<<<(n + 255) / 256, 256, 0, stream>>>(s, d, n);
  };
  cv(xp_w0, h_xpw0, 512);   cv(xp_w1, h_xpw1, 4096);
  cv(xp_w2, h_xpw2, 4096);  cv(xp_w3, h_xpw3, 4096);
  cv(xp_w4, h_xpw4, 4096);  cv(xpp_w0, h_xppw0, 4096);
  cv(xpp_w1, h_xppw1, 4096);
  cv(net_w0, h_nw0, 512);   cv(net_w1, h_nw1, 512);
  cv(lo_w0, h_low0, 64);    cv(lo_w1, h_low1, 512);
  cv(lc_w0, h_lcw0, 64);    cv(lc_w1, h_lcw1, 4096);
  cv(lz_w0, h_lzw0, 4096);  cv(lz_w1, h_lzw1, 512);

  // x_rec = x
  const int nrec = Bsz * Lsz * CINsz;
  copy_f32_kernel<<<(nrec + 255) / 256, 256, 0, stream>>>(x, out, nrec);

  // x_old = x_proj(x)
  xproj_kernel<<<(Bsz * Lsz) / 64, 128, 0, stream>>>(
      x, h_xpw0, xp_b0, h_xpw1, xp_b1, h_xpw2, xp_b2, h_xpw3, xp_b3, h_xpw4,
      xp_b4, xold);

  // a0 = comp(silu(conv(x_old)))
  conv_comp_kernel<<<Bsz, 256, 0, stream>>>(xold, conv_w, conv_b, comp_w,
                                            comp_b, a0b);

  // v (Householder direction)
  v_kernel<<<Bsz / 64, 128, 0, stream>>>(xold, h_xppw0, xpp_b0, h_xppw1, xpp_b1,
                                         vbuf);

  // recurrence -> trajectory
  scan_kernel<<<NTILES_B, 32, 0, stream>>>(a0b, xold, vbuf, h_nw0, net_b0,
                                           h_nw1, net_b1, h_low0, lo_b0, h_low1,
                                           lo_b1, span_A, atraj, ztraj);

  // decoders -> x_pred (fully parallel over P*B)
  decoder_kernel<<<(Psz * NTILES_B) / 2, 64, 0, stream>>>(
      atraj, ztraj, h_lcw0, lc_b0, h_lcw1, lc_b1, h_lzw0, lz_b0, h_lzw1, lz_b1,
      out + (size_t)nrec);
}
